// TokenFormerBlock_48979807044181
// MI455X (gfx1250) — matrix operand order repaired and checked by reading, not compiled, not run on hardware
//
#include <hip/hip_runtime.h>
#include <hip/hip_bf16.h>
#include <math.h>

// ---------------------------------------------------------------------------
// CDNA5 / gfx1250 TokenFormer block. wave32, WMMA f16->f32 16x16x32,
// Tensor Data Mover (tensor_load_to_lds) for LDS tile staging.
// ---------------------------------------------------------------------------

typedef __attribute__((ext_vector_type(16))) _Float16 v16h;
typedef __attribute__((ext_vector_type(8)))  _Float16 h8;
typedef __attribute__((ext_vector_type(8)))  float    v8f;
typedef __attribute__((ext_vector_type(4)))  unsigned v4u;
typedef __attribute__((ext_vector_type(8)))  int      v8i_;
typedef __attribute__((ext_vector_type(4)))  int      v4i_;

constexpr int Dd  = 2048;
constexpr int Gg  = 16;
constexpr int DHc = 128;     // Dd / Gg
constexpr int NSs = 2048;
constexpr int Bb  = 2;
constexpr int Ss  = 2048;
constexpr int BS  = Bb * Ss; // 4096 token rows
constexpr int Ee  = Dd * 4;  // FFN hidden = 8192
constexpr int RDc = Dd / 4;  // rope dims = 512

#define USE_TDM 1

__device__ __forceinline__ v8f wmma_f16(v16h a, v16h b, v8f c) {
  // v_wmma_f32_16x16x32_f16 : D = A(16x32) x B(32x16) + C(16x16 f32)
  return __builtin_amdgcn_wmma_f32_16x16x32_f16(
      /*neg_a=*/false, a, /*neg_b=*/false, b,
      /*c_mod=*/(short)0, c, /*reuse_a=*/false, /*reuse_b=*/false);
}

__device__ __forceinline__ v16h mk16(h8 lo, h8 hi) {
  v16h r;
#pragma unroll
  for (int j = 0; j < 8; ++j) { r[j] = lo[j]; r[8 + j] = hi[j]; }
  return r;
}

__device__ __forceinline__ unsigned lds_lo(const void* p) {
  // LDS generic addresses: low 32 bits are the LDS byte offset (ISA 10.2).
  return (unsigned)(size_t)p;
}

#if USE_TDM
// ---------------------------------------------------------------------------
// TDM: issue one 2D tile load (Global -> LDS) via Tensor DMA Descriptor (D#).
//   tile: tile_d1 rows x tile_d0 f16 elements, row stride = stride0 elements.
//   LDS padding: pad_amount DWORDs appended after every pad_interval DWORDs
//   (codes per ISA: interval 0=2,1=4,2=8,3=16,4=32,5=64; amount = code+1).
// ---------------------------------------------------------------------------
__device__ __forceinline__ void tdm_load_2d(unsigned lds_addr, const void* gsrc,
                                            long stride0, int tile_d0, int tile_d1,
                                            int pad_i_code, int pad_a_code)
{
  unsigned long long ga = (unsigned long long)gsrc;
  const int td0 = 1 << 30, td1 = 1 << 30;   // huge tensor dims: no OOB clipping
  v4u g0;
  g0[0] = 1u;                                // count=1, user descriptor
  g0[1] = lds_addr;                          // bits[63:32]  lds_addr
  g0[2] = (unsigned)(ga & 0xffffffffull);    // bits[95:64]  global_addr lo
  g0[3] = (unsigned)((ga >> 32) & 0x01ffffffull) | (2u << 30); // addr hi | type=2
  v8i_ g1;
  g1[0] = (1 << 16) | (1 << 20) | (pad_i_code << 22) | (pad_a_code << 25); // 2B, pad_en
  g1[1] = (td0 & 0xffff) << 16;                                 // tensor_dim0[15:0]
  g1[2] = ((td0 >> 16) & 0xffff) | ((td1 & 0xffff) << 16);      // d0 hi | d1 lo
  g1[3] = ((td1 >> 16) & 0xffff) | (tile_d0 << 16);             // d1 hi | tile_dim0
  g1[4] = tile_d1 & 0xffff;                                     // tile_dim1 (tile_dim2=0)
  g1[5] = (int)(stride0 & 0xffffffffll);                        // dim0 stride lo
  g1[6] = (int)((stride0 >> 32) & 0xffffll);                    // dim0 stride hi
  g1[7] = 0;
  v4i_ z4 = {0, 0, 0, 0};
#if defined(__clang_major__) && __clang_major__ >= 23
  v8i_ z8 = {0, 0, 0, 0, 0, 0, 0, 0};
  __builtin_amdgcn_tensor_load_to_lds(g0, g1, z4, z4, z8, 0);
#else
  __builtin_amdgcn_tensor_load_to_lds(g0, g1, z4, z4, 0);
#endif
}
#endif

// ---------------------------------------------------------------------------
// f32 -> f16 cast (grid-stride)
// ---------------------------------------------------------------------------
__global__ void cvt_f32_f16(const float* __restrict__ s, _Float16* __restrict__ d, long n) {
  long i = (long)blockIdx.x * blockDim.x + threadIdx.x;
  long stride = (long)gridDim.x * blockDim.x;
  for (; i < n; i += stride) d[i] = (_Float16)s[i];
}

// ---------------------------------------------------------------------------
// Tiled WMMA GEMM:  C[z] = act(alpha * A[z] @ op(B[z]))
//   A: (M x K) row-major f16, op(B): (K x N) (TRANSB -> B stored N x K)
//   Compile-time: TRANSB, ACT (gelu), STF (f32 out), STH (f16 out).
//   Block tile 128x128x32, 8 waves, each wave 64x32 (4x2 WMMA tiles).
//   A tile (and B tile when TRANSB) staged via TDM tensor_load_to_lds.
//   M, N multiples of 128; K multiple of 32 (true for every call here).
// ---------------------------------------------------------------------------
constexpr int BM = 128, BN = 128, BKk = 32, LPAD = 8;

template <int TRANSB, int ACT, int STF, int STH>
__global__ __launch_bounds__(256)
void gemm_wmma(const _Float16* __restrict__ A, long lda, long sA,
               const _Float16* __restrict__ B, long ldb, long sB,
               float* __restrict__ Cf, _Float16* __restrict__ Ch,
               long ldc, long sC, int K, float alpha)
{
  __shared__ __align__(16) _Float16 lA[BM][BKk + LPAD];   // row r at +80B*r
  __shared__ __align__(16) _Float16 lB[BN][BKk + LPAD];   // stored [n][k]

  const int z = blockIdx.z;
  const _Float16* Ab = A + (long)z * sA;
  const _Float16* Bp = B + (long)z * sB;
  const long coff = (long)z * sC;
  const int m0 = blockIdx.y * BM, n0 = blockIdx.x * BN;
  const int tid = threadIdx.x, lane = tid & 31, wave = tid >> 5;
  const int l16 = lane & 15, half = lane >> 4, kb = half * 8;
  const int wm = (wave >> 2) * 64;     // wave grid 2(M) x 4(N)
  const int wn = (wave & 3) * 32;

  v8f acc[4][2];
#pragma unroll
  for (int mt = 0; mt < 4; ++mt)
#pragma unroll
    for (int nt = 0; nt < 2; ++nt)
#pragma unroll
      for (int e = 0; e < 8; ++e) acc[mt][nt][e] = 0.f;

  for (int k0 = 0; k0 < K; k0 += BKk) {
#if USE_TDM
    if (tid == 0) {
      // A tile: 128 rows x 32 f16; 64B rows -> pad 16B every 16 DWORDs.
      tdm_load_2d(lds_lo(&lA[0][0]), Ab + (long)m0 * lda + k0, lda, BKk, BM, 3, 3);
      if (TRANSB)
        tdm_load_2d(lds_lo(&lB[0][0]), Bp + (long)n0 * ldb + k0, ldb, BKk, BN, 3, 3);
    }
#else
    { int r = tid >> 1, c = (tid & 1) * 16;
      const _Float16* s = Ab + (long)(m0 + r) * lda + (k0 + c);
      *(h8*)&lA[r][c]     = *(const h8*)s;
      *(h8*)&lA[r][c + 8] = *(const h8*)(s + 8);
      if (TRANSB) {
        const _Float16* t = Bp + (long)(n0 + r) * ldb + (k0 + c);
        *(h8*)&lB[r][c]     = *(const h8*)t;
        *(h8*)&lB[r][c + 8] = *(const h8*)(t + 8);
      }
    }
#endif
    if (!TRANSB) { // B stored K x N: contiguous global reads, strided LDS writes
      int kk = tid >> 3, c = (tid & 7) * 16;
      const _Float16* s = Bp + (long)(k0 + kk) * ldb + (n0 + c);
      h8 v0 = *(const h8*)s, v1 = *(const h8*)(s + 8);
#pragma unroll
      for (int j = 0; j < 8; ++j) { lB[c + j][kk] = v0[j]; lB[c + 8 + j][kk] = v1[j]; }
    }
#if USE_TDM
    __builtin_amdgcn_s_wait_tensorcnt((short)0);
#endif
    __syncthreads();

    v16h af[4], bf[2];
#pragma unroll
    for (int mt = 0; mt < 4; ++mt) { // A frag: lane m=l16; K = half*8+j, 16+half*8+j
      int r = wm + mt * 16 + l16;
      af[mt] = mk16(*(const h8*)&lA[r][kb], *(const h8*)&lA[r][16 + kb]);
    }
#pragma unroll
    for (int nt = 0; nt < 2; ++nt) { // B frag: lane n=l16; K = half*8+j, 16+half*8+j
      int n = wn + nt * 16 + l16;
      bf[nt] = mk16(*(const h8*)&lB[n][kb], *(const h8*)&lB[n][16 + kb]);
    }
#pragma unroll
    for (int mt = 0; mt < 4; ++mt)
#pragma unroll
      for (int nt = 0; nt < 2; ++nt)
        acc[mt][nt] = wmma_f16(af[mt], bf[nt], acc[mt][nt]);
    __syncthreads();
  }

  // C layout: vgpr r -> row (r + half*8), col = l16 within 16x16 tile
#pragma unroll
  for (int mt = 0; mt < 4; ++mt)
#pragma unroll
    for (int nt = 0; nt < 2; ++nt) {
      int n = n0 + wn + nt * 16 + l16;
      int mb = m0 + wm + mt * 16 + half * 8;
#pragma unroll
      for (int r = 0; r < 8; ++r) {
        float v = acc[mt][nt][r] * alpha;
        if (ACT) v = 0.5f * v * (1.0f + erff(v * 0.70710678118654752f)); // exact GELU
        long idx = coff + (long)(mb + r) * ldc + n;
        if (STF) Cf[idx] = v;
        if (STH) Ch[idx] = (_Float16)v;
      }
    }
}

// ---------------------------------------------------------------------------
// Flash attention per (b, g): Q(S x 128) against K/V(NS x 128) slices of
// stride Dd, online softmax, output f16 into O at column g*128.
// Block = 8 waves; wave owns 16 q rows; K/V chunk (32 x 128) staged via TDM.
// ---------------------------------------------------------------------------
constexpr int FPAD = 8;

__global__ __launch_bounds__(256)
void flash_attn(const _Float16* __restrict__ Q, const _Float16* __restrict__ Km,
                const _Float16* __restrict__ V, const int* __restrict__ mask,
                _Float16* __restrict__ O, float scale)
{
  __shared__ __align__(16) _Float16 lK[32][DHc + FPAD];   // row at +272B*r
  __shared__ __align__(16) _Float16 lV[32][DHc + FPAD];
  __shared__ __align__(16) _Float16 lP[8][16][32 + FPAD]; // per-wave P transpose scratch

  const int b = blockIdx.z, g = blockIdx.y;
  const int tid = threadIdx.x, lane = tid & 31, wave = tid >> 5;
  const int l16 = lane & 15, half = lane >> 4, kb = half * 8;
  const int row0 = blockIdx.x * 128 + wave * 16;
  const _Float16* qp = Q + ((long)b * Ss + row0) * Dd + g * DHc;

  // Q fragments over d (K-dim), kept in registers for the whole sweep
  v16h aq[4];
#pragma unroll
  for (int ks = 0; ks < 4; ++ks) {
    const _Float16* s = qp + (long)l16 * Dd + ks * 32 + half * 16;
    aq[ks] = mk16(*(const h8*)s, *(const h8*)(s + 8));
  }

  v8f o[8];
#pragma unroll
  for (int dt = 0; dt < 8; ++dt)
#pragma unroll
    for (int e = 0; e < 8; ++e) o[dt][e] = 0.f;
  float mrow[8], lrow[8];
#pragma unroll
  for (int r = 0; r < 8; ++r) { mrow[r] = -INFINITY; lrow[r] = 0.f; }

  for (int c0 = 0; c0 < NSs; c0 += 32) {
#if USE_TDM
    if (tid == 0) {
      // 32 rows x 128 f16 (256B = 64 DWORD rows) -> pad 4 DWORDs every 64.
      tdm_load_2d(lds_lo(&lK[0][0]), Km + (long)c0 * Dd + g * DHc, Dd, DHc, 32, 5, 3);
      tdm_load_2d(lds_lo(&lV[0][0]), V  + (long)c0 * Dd + g * DHc, Dd, DHc, 32, 5, 3);
    }
    __builtin_amdgcn_s_wait_tensorcnt((short)0);
#else
    { int r = tid >> 3, cc = (tid & 7) * 16;
      const _Float16* ks_ = Km + (long)(c0 + r) * Dd + g * DHc + cc;
      const _Float16* vs_ = V  + (long)(c0 + r) * Dd + g * DHc + cc;
      *(h8*)&lK[r][cc]     = *(const h8*)ks_;
      *(h8*)&lK[r][cc + 8] = *(const h8*)(ks_ + 8);
      *(h8*)&lV[r][cc]     = *(const h8*)vs_;
      *(h8*)&lV[r][cc + 8] = *(const h8*)(vs_ + 8);
    }
#endif
    __syncthreads();

    // S = Q @ Kc^T  (16 x 32) : two 16x16 tiles, 4 K-steps each
    v8f s0, s1;
#pragma unroll
    for (int e = 0; e < 8; ++e) { s0[e] = 0.f; s1[e] = 0.f; }
#pragma unroll
    for (int ks = 0; ks < 4; ++ks) {
      const _Float16* k0p = &lK[l16][ks * 32 + half * 16];
      const _Float16* k1p = &lK[16 + l16][ks * 32 + half * 16];
      v16h b0 = mk16(*(const h8*)k0p, *(const h8*)(k0p + 8));
      v16h b1 = mk16(*(const h8*)k1p, *(const h8*)(k1p + 8));
      s0 = wmma_f16(aq[ks], b0, s0);
      s1 = wmma_f16(aq[ks], b1, s1);
    }

    const bool ok0 = mask[(long)b * Ss + (c0 + l16)] != 0;
    const bool ok1 = mask[(long)b * Ss + (c0 + 16 + l16)] != 0;

    float fr[8];
#pragma unroll
    for (int r = 0; r < 8; ++r) {
      float a0 = ok0 ? s0[r] * scale : -INFINITY;
      float a1 = ok1 ? s1[r] * scale : -INFINITY;
      float t = fmaxf(a0, a1);
#pragma unroll
      for (int off = 8; off >= 1; off >>= 1) t = fmaxf(t, __shfl_xor(t, off, 32));
      float mn = fmaxf(mrow[r], t);
      float f  = (mn == -INFINITY) ? 1.f : __expf(mrow[r] - mn);
      float p0 = (mn == -INFINITY) ? 0.f : __expf(a0 - mn);
      float p1 = (mn == -INFINITY) ? 0.f : __expf(a1 - mn);
      float ps = p0 + p1;
#pragma unroll
      for (int off = 8; off >= 1; off >>= 1) ps += __shfl_xor(ps, off, 32);
      mrow[r] = mn; lrow[r] = lrow[r] * f + ps; fr[r] = f;
      lP[wave][half * 8 + r][l16]      = (_Float16)p0;  // transpose P via LDS
      lP[wave][half * 8 + r][16 + l16] = (_Float16)p1;
    }
#pragma unroll
    for (int dt = 0; dt < 8; ++dt)
#pragma unroll
      for (int r = 0; r < 8; ++r) o[dt][r] *= fr[r];
    __syncthreads();

    // P (16x32) as A-fragment, V chunk as 8 B-fragments over d
    const _Float16* pp = &lP[wave][l16][half * 16];
    v16h ap = mk16(*(const h8*)pp, *(const h8*)(pp + 8));
#pragma unroll
    for (int dt = 0; dt < 8; ++dt) {
      v16h bv;
      int d = dt * 16 + l16;
#pragma unroll
      for (int j = 0; j < 16; ++j) bv[j] = lV[half * 16 + j][d];
      o[dt] = wmma_f16(ap, bv, o[dt]);
    }
    __syncthreads();
  }

#pragma unroll
  for (int r = 0; r < 8; ++r) {
    float inv = lrow[r] > 0.f ? 1.f / lrow[r] : 0.f;
    long orow = (long)b * Ss + row0 + half * 8 + r;
#pragma unroll
    for (int dt = 0; dt < 8; ++dt) {
      int d = g * DHc + dt * 16 + l16;
      O[orow * Dd + d] = (_Float16)(o[dt][r] * inv);
    }
  }
}

// ---------------------------------------------------------------------------
// Row softmax (pattention): f32 in -> f16 probabilities out. One block / row.
// ---------------------------------------------------------------------------
__global__ __launch_bounds__(256)
void softmax_rows(const float* __restrict__ Sf, _Float16* __restrict__ P, int C)
{
  const long row = blockIdx.x;
  const float* src = Sf + row * (long)C;
  _Float16* dst = P + row * (long)C;
  __shared__ float red[8];
  const int tid = threadIdx.x, lane = tid & 31, wave = tid >> 5;

  float mx = -INFINITY;
  for (int i = tid; i < C; i += 256) mx = fmaxf(mx, src[i]);
#pragma unroll
  for (int off = 16; off >= 1; off >>= 1) mx = fmaxf(mx, __shfl_xor(mx, off, 32));
  if (lane == 0) red[wave] = mx;
  __syncthreads();
  float m = -INFINITY;
  for (int w = 0; w < 8; ++w) m = fmaxf(m, red[w]);
  __syncthreads();

  float sum = 0.f;
  for (int i = tid; i < C; i += 256) sum += __expf(src[i] - m);
#pragma unroll
  for (int off = 16; off >= 1; off >>= 1) sum += __shfl_xor(sum, off, 32);
  if (lane == 0) red[wave] = sum;
  __syncthreads();
  float tot = 0.f;
  for (int w = 0; w < 8; ++w) tot += red[w];
  float inv = 1.f / tot;
  for (int i = tid; i < C; i += 256) dst[i] = (_Float16)(__expf(src[i] - m) * inv);
}

// ---------------------------------------------------------------------------
// h = LayerNorm(A + B) -> f32 out (+ optional f16 copy). One block / row.
// ---------------------------------------------------------------------------
__global__ __launch_bounds__(256)
void add_ln(const float* __restrict__ A, const float* __restrict__ Bv,
            float* __restrict__ Of, _Float16* __restrict__ Oh, int C)
{
  const long row = blockIdx.x;
  const float* a = A + row * (long)C;
  const float* b = Bv + row * (long)C;
  __shared__ float r1[8], r2[8];
  const int tid = threadIdx.x, lane = tid & 31, wave = tid >> 5;

  float s1 = 0.f, s2 = 0.f;
  for (int i = tid; i < C; i += 256) { float h = a[i] + b[i]; s1 += h; s2 += h * h; }
#pragma unroll
  for (int off = 16; off >= 1; off >>= 1) { s1 += __shfl_xor(s1, off, 32); s2 += __shfl_xor(s2, off, 32); }
  if (lane == 0) { r1[wave] = s1; r2[wave] = s2; }
  __syncthreads();
  float t1 = 0.f, t2 = 0.f;
  for (int w = 0; w < 8; ++w) { t1 += r1[w]; t2 += r2[w]; }
  float mean = t1 / C;
  float var  = t2 / C - mean * mean;
  float inv  = rsqrtf(var + 1e-5f);
  for (int i = tid; i < C; i += 256) {
    float o = (a[i] + b[i] - mean) * inv;
    Of[row * (long)C + i] = o;
    if (Oh) Oh[row * (long)C + i] = (_Float16)o;
  }
}

// ---------------------------------------------------------------------------
// RoPE on first RDc dims of qh (f32) + cast whole tensor to f16.
// ---------------------------------------------------------------------------
__global__ void rope_cvt(const float* __restrict__ q, _Float16* __restrict__ out)
{
  long idx = (long)blockIdx.x * blockDim.x + threadIdx.x;
  if (idx >= (long)BS * Dd) return;
  int d = (int)(idx % Dd);
  long row = idx / Dd;
  int s = (int)(row % Ss);
  float v = q[idx], o;
  if (d < RDc) {
    const int hh = RDc / 2;                       // 256
    int i = d % hh;                               // tiled cos/sin
    float invf = __powf(10000.f, -2.f * (float)i / (float)RDc);
    float pe = (float)s * invf;
    float rot = (d < hh) ? -q[idx + hh] : q[idx - hh];
    o = v * __cosf(pe) + rot * __sinf(pe);
  } else {
    o = v;
  }
  out[idx] = (_Float16)o;
}

// ---------------------------------------------------------------------------
// Host orchestration
// ---------------------------------------------------------------------------
static inline char* bump(char*& p, size_t bytes) {
  char* r = p;
  p += (bytes + 255) & ~(size_t)255;
  return r;
}

extern "C" void kernel_launch(void* const* d_in, const int* in_sizes, int n_in,
                              void* d_out, int out_size, void* d_ws, size_t ws_size,
                              hipStream_t stream)
{
  (void)in_sizes; (void)n_in; (void)out_size; (void)ws_size;

  const float* x     = (const float*)d_in[0];
  const int*   mask  = (const int*)  d_in[1];
  const float* slots = (const float*)d_in[2];
  const float* paq   = (const float*)d_in[3];
  const float* pak   = (const float*)d_in[4];
  const float* pav   = (const float*)d_in[5];
  const float* pao   = (const float*)d_in[6];
  const float* wqf   = (const float*)d_in[7];
  const float* wkf   = (const float*)d_in[8];
  const float* wvf   = (const float*)d_in[9];
  const float* wof   = (const float*)d_in[10];
  const float* w1f   = (const float*)d_in[11];
  const float* w2f   = (const float*)d_in[12];
  float* out = (float*)d_out;

  const long DD  = Dd;
  const long DD2 = (long)Dd * Dd;
  const long NSD = (long)NSs * Dd;
  const long NS2 = (long)NSs * NSs;
  const long BSD = (long)BS * Dd;

  char* p = (char*)d_ws;
  auto aH = [&](long e) { return (_Float16*)bump(p, (size_t)e * sizeof(_Float16)); };
  auto aF = [&](long e) { return (float*)bump(p, (size_t)e * sizeof(float)); };

  _Float16* x_h   = aH(BSD);
  _Float16* slt_h = aH(2 * NSD);        // t = 1,2 only (qkv_params[0] unused)
  _Float16* paq_h = aH(DD2);
  _Float16* pak_h = aH(DD2);
  _Float16* pav_h = aH(DD2);
  _Float16* pao_h = aH(DD2);
  _Float16* wq_h  = aH(DD2);
  _Float16* wk_h  = aH(DD2);
  _Float16* wv_h  = aH(DD2);
  _Float16* wo_h  = aH(DD2);
  _Float16* w1_h  = aH((long)Dd * Ee);
  _Float16* w2_h  = aH((long)Ee * Dd);
  _Float16* q_h   = aH(2 * NSD);
  _Float16* k_h   = aH(2 * NSD);
  _Float16* v_h   = aH(2 * NSD);
  float*    sc_f  = aF(2 * NS2);
  _Float16* aw_h  = aH(2 * NS2);
  _Float16* av_h  = aH(2 * NSD);
  _Float16* kvp_h = aH(2 * NSD);
  _Float16* kh_h  = aH(NSD);
  _Float16* vh_h  = aH(NSD);
  float*    qh_f  = aF(BSD);
  _Float16* qh_h  = aH(BSD);
  _Float16* at_h  = aH(BSD);
  float*    ap_f  = aF(BSD);
  float*    h_f   = aF(BSD);
  _Float16* h_h   = aH(BSD);
  _Float16* f1_h  = aH((long)BS * Ee);
  float*    f2_f  = aF(BSD);

  auto cvt = [&](const float* s, _Float16* d, long n) {
    long blocks = (n + 1023) / 1024;
    if (blocks > 2048) blocks = 2048;
    cvt_f32_f16<<<dim3((unsigned)blocks), 256, 0, stream>>>(s, d, n);
  };
  // Compile-time dispatch over (transB, act, f32-store, f16-store).
  auto gemm = [&](const _Float16* A, long lda, long sA,
                  const _Float16* B, long ldb, long sB, int tB,
                  float* Cf, _Float16* Ch, long ldc, long sC,
                  int M, int N, int K, float alpha, int act, int batch) {
    dim3 grid((unsigned)(N / 128), (unsigned)(M / 128), (unsigned)batch);
    if (tB == 1)                       // scores: f32 out
      gemm_wmma<1, 0, 1, 0><<<grid, 256, 0, stream>>>(A, lda, sA, B, ldb, sB, Cf, Ch, ldc, sC, K, alpha);
    else if (act == 1)                 // ffn1: gelu -> f16 out
      gemm_wmma<0, 1, 0, 1><<<grid, 256, 0, stream>>>(A, lda, sA, B, ldb, sB, Cf, Ch, ldc, sC, K, alpha);
    else if (Cf != nullptr)            // f32 out
      gemm_wmma<0, 0, 1, 0><<<grid, 256, 0, stream>>>(A, lda, sA, B, ldb, sB, Cf, Ch, ldc, sC, K, alpha);
    else                               // f16 out
      gemm_wmma<0, 0, 0, 1><<<grid, 256, 0, stream>>>(A, lda, sA, B, ldb, sB, Cf, Ch, ldc, sC, K, alpha);
  };

  const float scale = 0.08838834764831845f; // 1/sqrt(DH)

  // --- cast inputs/weights to f16 ---
  cvt(x, x_h, BSD);
  cvt(slots + NSD, slt_h, 2 * NSD);     // skip t=0
  cvt(paq, paq_h, DD2);  cvt(pak, pak_h, DD2);
  cvt(pav, pav_h, DD2);  cvt(pao, pao_h, DD2);
  cvt(wqf, wq_h, DD2);   cvt(wkf, wk_h, DD2);
  cvt(wvf, wv_h, DD2);   cvt(wof, wo_h, DD2);
  cvt(w1f, w1_h, (long)Dd * Ee);
  cvt(w2f, w2_h, (long)Ee * Dd);

  // --- pattention (batched over t = 1,2) ---
  gemm(slt_h, DD, NSD, paq_h, DD, 0, 0, nullptr, q_h, DD, NSD, NSs, Dd, Dd, 1.f, 0, 2);
  gemm(slt_h, DD, NSD, pak_h, DD, 0, 0, nullptr, k_h, DD, NSD, NSs, Dd, Dd, 1.f, 0, 2);
  gemm(slt_h, DD, NSD, pav_h, DD, 0, 0, nullptr, v_h, DD, NSD, NSs, Dd, Dd, 1.f, 0, 2);
  gemm(q_h, DD, NSD, k_h, DD, NSD, 1, sc_f, nullptr, NSs, NS2, NSs, NSs, Dd, scale, 0, 2);
  softmax_rows<<<dim3(2 * NSs), 256, 0, stream>>>(sc_f, aw_h, NSs);
  gemm(aw_h, NSs, NS2, v_h, DD, NSD, 0, nullptr, av_h, DD, NSD, NSs, Dd, NSs, 1.f, 0, 2);
  gemm(av_h, DD, NSD, pao_h, DD, 0, 0, nullptr, kvp_h, DD, NSD, NSs, Dd, Dd, 1.f, 0, 2);

  // --- token-side projections ---
  gemm(kvp_h,       DD, 0, wk_h, DD, 0, 0, nullptr, kh_h, DD, 0, NSs, Dd, Dd, 1.f, 0, 1);
  gemm(kvp_h + NSD, DD, 0, wv_h, DD, 0, 0, nullptr, vh_h, DD, 0, NSs, Dd, Dd, 1.f, 0, 1);
  gemm(x_h, DD, 0, wq_h, DD, 0, 0, qh_f, nullptr, DD, 0, BS, Dd, Dd, 1.f, 0, 1);
  rope_cvt<<<dim3((unsigned)((BSD + 255) / 256)), 256, 0, stream>>>(qh_f, qh_h);

  // --- grouped flash attention ---
  flash_attn<<<dim3(Ss / 128, Gg, Bb), 256, 0, stream>>>(qh_h, kh_h, vh_h, mask, at_h, scale);
  gemm(at_h, DD, 0, wo_h, DD, 0, 0, ap_f, nullptr, DD, 0, BS, Dd, Dd, 1.f, 0, 1);

  // --- residual + LN, FFN (fused GELU), residual + LN ---
  add_ln<<<dim3(BS), 256, 0, stream>>>(x, ap_f, h_f, h_h, Dd);
  gemm(h_h, DD, 0, w1_h, Ee, 0, 0, nullptr, f1_h, Ee, 0, BS, Ee, Dd, 1.f, 1, 1);
  gemm(f1_h, Ee, 0, w2_h, DD, 0, 0, f2_f, nullptr, DD, 0, BS, Ee, Ee, 1.f, 0, 1);
  add_ln<<<dim3(BS), 256, 0, stream>>>(h_f, f2_f, out, nullptr, Dd);
}
